// NameGCN_10479720202575
// MI455X (gfx1250) — matrix operand order repaired and checked by reading, not compiled, not run on hardware
//
#include <hip/hip_runtime.h>

#define N_NODES 100000
#define N_EDGES 1600000
#define DIM     128
#define N_SEEDS 15000

typedef __attribute__((ext_vector_type(16))) __bf16 v16bf;
typedef __attribute__((ext_vector_type(8)))  __bf16 v8bf;
typedef __attribute__((ext_vector_type(8)))  float  v8f;
typedef __attribute__((ext_vector_type(4)))  int    v4i;

#if __has_builtin(__builtin_amdgcn_global_load_async_to_lds_b128)
#define USE_ASYNC_LDS 1
#else
#define USE_ASYNC_LDS 0
#endif

// ---------- bf16 helpers (RNE split of f32 into hi+lo bf16) ----------
static __device__ __forceinline__ __bf16 f32_to_bf16_rne(float f) {
    union { float f; unsigned u; } v; v.f = f;
    unsigned r = v.u + 0x7FFFu + ((v.u >> 16) & 1u);   // round-to-nearest-even
    unsigned short s = (unsigned short)(r >> 16);
    __bf16 b; __builtin_memcpy(&b, &s, 2); return b;
}
static __device__ __forceinline__ float bf16_to_f32(__bf16 b) {
    unsigned short s; __builtin_memcpy(&s, &b, 2);
    union { unsigned u; float f; } v; v.u = ((unsigned)s) << 16; return v.f;
}

// ---------- split W (128x128 f32) into hi/lo bf16, row-major ----------
__global__ void w_split_kernel(const float* __restrict__ W,
                               __bf16* __restrict__ Whi, __bf16* __restrict__ Wlo) {
    int i = blockIdx.x * blockDim.x + threadIdx.x;
    if (i >= DIM * DIM) return;
    float w  = W[i];
    __bf16 h = f32_to_bf16_rne(w);
    Whi[i] = h;
    Wlo[i] = f32_to_bf16_rne(w - bf16_to_f32(h));
}

// ---------- stream split: x (f32) -> Xhi/Xlo (bf16), optional fused ReLU ----------
template <bool RELU>
__global__ void x_split_kernel(const float* __restrict__ x,
                               __bf16* __restrict__ Xhi, __bf16* __restrict__ Xlo,
                               int n4) {
    int i = blockIdx.x * blockDim.x + threadIdx.x;
    if (i >= n4) return;
    float4 v = ((const float4*)x)[i];
    if (RELU) {
        v.x = fmaxf(v.x, 0.f); v.y = fmaxf(v.y, 0.f);
        v.z = fmaxf(v.z, 0.f); v.w = fmaxf(v.w, 0.f);
    }
    __bf16 h0 = f32_to_bf16_rne(v.x), h1 = f32_to_bf16_rne(v.y);
    __bf16 h2 = f32_to_bf16_rne(v.z), h3 = f32_to_bf16_rne(v.w);
    __bf16 hv4[4] = { h0, h1, h2, h3 };
    __bf16 lv4[4] = { f32_to_bf16_rne(v.x - bf16_to_f32(h0)),
                      f32_to_bf16_rne(v.y - bf16_to_f32(h1)),
                      f32_to_bf16_rne(v.z - bf16_to_f32(h2)),
                      f32_to_bf16_rne(v.w - bf16_to_f32(h3)) };
    *(double*)(Xhi + (size_t)i * 4) = *(double*)hv4;   // 8B store
    *(double*)(Xlo + (size_t)i * 4) = *(double*)lv4;
}

// ---------- degree: init (self-loop), accumulate, rsqrt in place ----------
__global__ void deg_init_kernel(float* __restrict__ deg) {
    int i = blockIdx.x * blockDim.x + threadIdx.x;
    if (i < N_NODES) deg[i] = 1.0f;                     // self-loop
}
__global__ void deg_accum_kernel(const int* __restrict__ dst, float* __restrict__ deg) {
    int e = blockIdx.x * blockDim.x + threadIdx.x;
    if (e < N_EDGES) atomicAdd(&deg[dst[e]], 1.0f);
}
__global__ void deg_rsqrt_kernel(float* __restrict__ deg) {
    int i = blockIdx.x * blockDim.x + threadIdx.x;
    if (i < N_NODES) deg[i] = rsqrtf(deg[i]);           // deg -> dinv in place
}

// ---------- WMMA GEMM: h = x @ W  (bf16x3 split, f32 accumulate) ----------
// A comes pre-split (Xhi/Xlo bf16); tile staged to LDS via async-to-LDS.
// Also writes agg_init = (1 + dinv^2) * h (residual + self-loop message).
// Grid: N_NODES/16 blocks x 256 threads (8 waves); wave w owns N cols [16w,16w+16).
#define ASTRIDE 136   // bf16 elements per LDS row (16B aligned, bank-spread)
__global__ __launch_bounds__(256) void gcn_gemm_kernel(
        const __bf16* __restrict__ Xhi, const __bf16* __restrict__ Xlo,
        const __bf16* __restrict__ Whi, const __bf16* __restrict__ Wlo,
        const float*  __restrict__ dinv,
        float* __restrict__ h, float* __restrict__ agg) {
    __shared__ __bf16 ahi_s[16 * ASTRIDE];
    __shared__ __bf16 alo_s[16 * ASTRIDE];
    const int m0  = blockIdx.x * 16;
    const int tid = threadIdx.x;

    // ---- stage 16x128 bf16 tiles (hi & lo): 256 chunks of 16B each ----
    {
        const int r  = tid >> 4;            // row 0..15
        const int cc = tid & 15;            // 16B chunk within row
        const size_t goff = (size_t)(m0 + r) * DIM + cc * 8;  // bf16 elements
        const int    loff = r * ASTRIDE + cc * 8;
#if USE_ASYNC_LDS
        __builtin_amdgcn_global_load_async_to_lds_b128(
            (v4i*)(Xhi + goff), (v4i*)(ahi_s + loff), 0, 0);
        __builtin_amdgcn_global_load_async_to_lds_b128(
            (v4i*)(Xlo + goff), (v4i*)(alo_s + loff), 0, 0);
#if __has_builtin(__builtin_amdgcn_s_wait_asynccnt)
        __builtin_amdgcn_s_wait_asynccnt(0);
#else
        asm volatile("s_wait_asynccnt 0" ::: "memory");
#endif
#else
        *(v8bf*)(ahi_s + loff) = *(const v8bf*)(Xhi + goff);
        *(v8bf*)(alo_s + loff) = *(const v8bf*)(Xlo + goff);
#endif
    }
    __syncthreads();

    const int wave = tid >> 5;
    const int lane = tid & 31;
    const int nw   = wave * 16;
    const int hi8  = (lane >> 4) * 8;                 // K offset for lanes 16..31
    const __bf16* arow_h = &ahi_s[(lane & 15) * ASTRIDE];
    const __bf16* arow_l = &alo_s[(lane & 15) * ASTRIDE];

    v8f acc = {};
    #pragma unroll
    for (int ks = 0; ks < 4; ++ks) {
        const int kb = ks * 32 + hi8;
        // A fragment per ISA 16-bit A layout: elems 0..7 = K kb..kb+7,
        // elems 8..15 = K kb+16..kb+23 (two aligned 16B LDS loads + shuffle)
        v8bf a0h = *(const v8bf*)(arow_h + kb);
        v8bf a1h = *(const v8bf*)(arow_h + kb + 16);
        v8bf a0l = *(const v8bf*)(arow_l + kb);
        v8bf a1l = *(const v8bf*)(arow_l + kb + 16);
        v16bf ahi = __builtin_shufflevector(a0h, a1h,
                        0,1,2,3,4,5,6,7,8,9,10,11,12,13,14,15);
        v16bf alo = __builtin_shufflevector(a0l, a1l,
                        0,1,2,3,4,5,6,7,8,9,10,11,12,13,14,15);
        // B fragment: lane L holds col nw+(L&15), K rows in the same order as A
        v16bf bhi, blo;
        #pragma unroll
        for (int s = 0; s < 16; ++s) {
            const int kr = kb + (s < 8 ? s : s + 8);
            bhi[s] = Whi[(size_t)kr * DIM + nw + (lane & 15)];
            blo[s] = Wlo[(size_t)kr * DIM + nw + (lane & 15)];
        }
        acc = __builtin_amdgcn_wmma_f32_16x16x32_bf16(false, ahi, false, bhi,
                                                      (short)0, acc, false, false);
        acc = __builtin_amdgcn_wmma_f32_16x16x32_bf16(false, alo, false, bhi,
                                                      (short)0, acc, false, false);
        acc = __builtin_amdgcn_wmma_f32_16x16x32_bf16(false, ahi, false, blo,
                                                      (short)0, acc, false, false);
    }

    // epilogue per 32-bit C/D layout: lane<16 -> (M=r, N=lane); lane>=16 -> (M=r+8)
    const int n  = nw + (lane & 15);
    const int mo = m0 + (lane >> 4) * 8;
    #pragma unroll
    for (int r = 0; r < 8; ++r) {
        int row  = mo + r;
        float v  = acc[r];
        float di = dinv[row];
        h[(size_t)row * DIM + n]   = v;
        agg[(size_t)row * DIM + n] = (1.0f + di * di) * v;   // residual + self-loop
    }
}

// ---------- edge scatter: agg[dst] += dinv[src]*dinv[dst] * h[src] ----------
// One wave per edge: lane covers 4 dims -> coalesced 512B row gather, 4 f32 atomics.
__global__ __launch_bounds__(256) void edge_scatter_kernel(
        const int* __restrict__ src, const int* __restrict__ dst,
        const float* __restrict__ dinv,
        const float* __restrict__ h, float* __restrict__ agg) {
    int t = blockIdx.x * blockDim.x + threadIdx.x;
    int e = t >> 5;
    if (e >= N_EDGES) return;
    int c = (t & 31) * 4;
    int s = src[e], d = dst[e];
    float nrm = dinv[s] * dinv[d];
    float4 hv = *(const float4*)(h + (size_t)s * DIM + c);
    float* ap = agg + (size_t)d * DIM + c;
    atomicAdd(ap + 0, nrm * hv.x);
    atomicAdd(ap + 1, nrm * hv.y);
    atomicAdd(ap + 2, nrm * hv.z);
    atomicAdd(ap + 3, nrm * hv.w);
}

// ---------- in-place ReLU (final layer output) ----------
__global__ void relu_kernel(float* __restrict__ a, int n4) {
    int i = blockIdx.x * blockDim.x + threadIdx.x;
    if (i >= n4) return;
    float4 v = ((float4*)a)[i];
    v.x = fmaxf(v.x, 0.f); v.y = fmaxf(v.y, 0.f);
    v.z = fmaxf(v.z, 0.f); v.w = fmaxf(v.w, 0.f);
    ((float4*)a)[i] = v;
}

// ---------- seed gather ----------
__global__ void seed_gather_kernel(const int* __restrict__ seeds,
                                   const float* __restrict__ ent,
                                   float* __restrict__ out) {
    int t = blockIdx.x * blockDim.x + threadIdx.x;
    int s = t >> 5;
    if (s >= N_SEEDS) return;
    int c = (t & 31) * 4;
    *(float4*)(out + (size_t)s * DIM + c) =
        *(const float4*)(ent + (size_t)seeds[s] * DIM + c);
}

extern "C" void kernel_launch(void* const* d_in, const int* in_sizes, int n_in,
                              void* d_out, int out_size, void* d_ws, size_t ws_size,
                              hipStream_t stream) {
    const int*   sr_seeds = (const int*)d_in[0];
    const int*   tg_seeds = (const int*)d_in[1];
    const int*   edges_sr = (const int*)d_in[2];
    const int*   edges_tg = (const int*)d_in[3];
    const float* emb_sr   = (const float*)d_in[4];
    const float* emb_tg   = (const float*)d_in[5];
    const float* W0       = (const float*)d_in[6];
    const float* W1       = (const float*)d_in[7];

    float* out         = (float*)d_out;
    float* sr_seed_out = out;
    float* tg_seed_out = out + (size_t)N_SEEDS * DIM;
    float* sr_ent_out  = out + (size_t)2 * N_SEEDS * DIM;
    float* tg_ent_out  = sr_ent_out + (size_t)N_NODES * DIM;

    // workspace carve
    char* ws = (char*)d_ws;
    float*  bufH   = (float*)ws;  ws += (size_t)N_NODES * DIM * sizeof(float);
    float*  bufAgg = (float*)ws;  ws += (size_t)N_NODES * DIM * sizeof(float);
    __bf16* Xhi    = (__bf16*)ws; ws += (size_t)N_NODES * DIM * 2;
    __bf16* Xlo    = (__bf16*)ws; ws += (size_t)N_NODES * DIM * 2;
    float*  dinv   = (float*)ws;  ws += (size_t)N_NODES * sizeof(float);
    __bf16* Whi0   = (__bf16*)ws; ws += (size_t)DIM * DIM * 2;
    __bf16* Wlo0   = (__bf16*)ws; ws += (size_t)DIM * DIM * 2;
    __bf16* Whi1   = (__bf16*)ws; ws += (size_t)DIM * DIM * 2;
    __bf16* Wlo1   = (__bf16*)ws; ws += (size_t)DIM * DIM * 2;

    const int nodeBlocks = (N_NODES + 255) / 256;
    const int edgeBlocks = (N_EDGES + 255) / 256;
    const int gemmBlocks = N_NODES / 16;                 // 6250, exact
    const int scatBlocks = (N_EDGES * 32 + 255) / 256;   // wave per edge
    const int n4         = N_NODES * DIM / 4;
    const int elemBlocks = (n4 + 255) / 256;
    const int seedBlocks = (N_SEEDS * 32 + 255) / 256;

    w_split_kernel<<<(DIM * DIM + 255) / 256, 256, 0, stream>>>(W0, Whi0, Wlo0);
    w_split_kernel<<<(DIM * DIM + 255) / 256, 256, 0, stream>>>(W1, Whi1, Wlo1);

    for (int g = 0; g < 2; ++g) {
        const int*   edges    = g ? edges_tg : edges_sr;
        const float* emb      = g ? emb_tg   : emb_sr;
        const int*   seeds    = g ? tg_seeds : sr_seeds;
        float*       ent_out  = g ? tg_ent_out  : sr_ent_out;
        float*       seed_out = g ? tg_seed_out : sr_seed_out;
        const int* src = edges;
        const int* dst = edges + N_EDGES;

        // symmetric normalization factors
        deg_init_kernel <<<nodeBlocks, 256, 0, stream>>>(dinv);
        deg_accum_kernel<<<edgeBlocks, 256, 0, stream>>>(dst, dinv);
        deg_rsqrt_kernel<<<nodeBlocks, 256, 0, stream>>>(dinv);

        // layer 1: split x0, GEMM -> bufH/bufAgg, scatter, fused relu+split -> x1
        x_split_kernel<false><<<elemBlocks, 256, 0, stream>>>(emb, Xhi, Xlo, n4);
        gcn_gemm_kernel     <<<gemmBlocks, 256, 0, stream>>>(Xhi, Xlo, Whi0, Wlo0,
                                                             dinv, bufH, bufAgg);
        edge_scatter_kernel <<<scatBlocks, 256, 0, stream>>>(src, dst, dinv, bufH, bufAgg);
        x_split_kernel<true><<<elemBlocks, 256, 0, stream>>>(bufAgg, Xhi, Xlo, n4);

        // layer 2: agg accumulates directly into d_out's ent region
        gcn_gemm_kernel     <<<gemmBlocks, 256, 0, stream>>>(Xhi, Xlo, Whi1, Wlo1,
                                                             dinv, bufH, ent_out);
        edge_scatter_kernel <<<scatBlocks, 256, 0, stream>>>(src, dst, dinv, bufH, ent_out);
        relu_kernel         <<<elemBlocks, 256, 0, stream>>>(ent_out, n4);

        seed_gather_kernel  <<<seedBlocks, 256, 0, stream>>>(seeds, ent_out, seed_out);
    }
}
